// LSTM_74045236183448
// MI455X (gfx1250) — matrix operand order repaired and checked by reading, not compiled, not run on hardware
//
#include <hip/hip_runtime.h>
#include <hip/hip_bf16.h>

typedef __attribute__((ext_vector_type(16))) _Float16 v16h;
typedef __attribute__((ext_vector_type(8)))  float    v8f;

#define B_TOT   4096
#define T_LEN   512
#define IN_F    3
#define H_F     128
#define NGATES  512      // 4*H
#define ROWS_WG 16
#define THREADS 256

// ---------------------------------------------------------------------------
// Prep: swizzle W_hh [512][128] f32 into f16 B-fragment layout for
// v_wmma_f32_16x16x32_f16.  B tile is 32(K) x 16(N); this code uses a
// contiguous-K B layout (not ISA table order): lanes 0-15 hold K=0..15 (N = lane), lanes 16-31 hold
// K=16..31 (N = lane-16), 16 f16 per lane.  Fragment order:
//   Bfrag[((n_tile*4 + k_chunk)*32 + lane)*16 + e]
// so the main kernel reads one contiguous v16h (32B) per lane; the compiler
// hoists all loop-invariant fragments into persistent VGPRs.
// ---------------------------------------------------------------------------
__global__ void prep_bfrag(const float* __restrict__ Whh,
                           _Float16* __restrict__ Bfrag) {
    int idx  = blockIdx.x * blockDim.x + threadIdx.x;  // 65536 total
    int e    = idx & 15;
    int lane = (idx >> 4) & 31;
    int kc   = (idx >> 9) & 3;
    int nt   = idx >> 11;                               // 0..31
    int n    = nt * 16 + (lane & 15);
    int k    = kc * 32 + (lane >> 4) * 16 + e;
    Bfrag[idx] = (_Float16)Whh[n * H_F + k];            // W_hh[n][k] = B(k,n)
}

// ---------------------------------------------------------------------------
// Main LSTM kernel.  One WG = 16 batch rows for all 512 timesteps; 256 WGs
// spread across WGPs.  8 waves each own 64 of the 512 gate columns
// (n_group = wave).  Per step each wave computes a 16x64 slice of
// gates = h(f16) @ W_hh.T via 16 WMMA ops (4 K-chunks x 4 N-tiles, B pinned
// in VGPRs), writes it to LDS, then all 256 threads do the elementwise LSTM
// update (c,h in registers) and scatter new h (f16) into A-fragment LDS.
// ---------------------------------------------------------------------------
__global__ __launch_bounds__(THREADS)
void lstm_wmma(const float* __restrict__ x,
               const _Float16* __restrict__ Bfrag,
               const float* __restrict__ Wih,
               const float* __restrict__ bih,
               const float* __restrict__ bhh,
               float* __restrict__ hout) {
    __shared__ float gatesLDS[ROWS_WG][NGATES + 1];          // 32.8 KB (padded)
    __shared__ __align__(32) _Float16 hA[4 * 32 * 16];       // 4 KB A-frags
    __shared__ float xs[ROWS_WG][IN_F];                      // staged x_t

    const int tid     = threadIdx.x;
    const int lane    = tid & 31;
    const int n_group = tid >> 5;        // wave 0..7 -> 64 gate cols each
    const int b0      = blockIdx.x * ROWS_WG;

    // Elementwise mapping: thread owns hidden col j for 8 rows (stride 2).
    const int j     = tid & (H_F - 1);
    const int rbase = tid >> 7;          // 0 or 1

    // Per-thread constants: W_ih rows and bias for the 4 gates of column j.
    float wih[4][3], bias[4];
#pragma unroll
    for (int g = 0; g < 4; ++g) {
        int col = g * H_F + j;
        wih[g][0] = Wih[col * 3 + 0];
        wih[g][1] = Wih[col * 3 + 1];
        wih[g][2] = Wih[col * 3 + 2];
        bias[g]   = bih[col] + bhh[col];
    }

    // h0 = 0 in A-fragment LDS; c0 = 0 in registers.
    for (int i = tid; i < 4 * 32 * 16; i += THREADS) hA[i] = (_Float16)0.f;
    float c_reg[8], h_reg[8];
#pragma unroll
    for (int k = 0; k < 8; ++k) { c_reg[k] = 0.f; h_reg[k] = 0.f; }

    const v16h* Bf = (const v16h*)Bfrag;   // L2-resident, shared by all WGs

#pragma unroll 1
    for (int t = 0; t < T_LEN; ++t) {
        __syncthreads();   // hA/xs from previous iteration fully consumed

        // Stage x_t for our 16 rows (48 floats) into LDS.
        if (tid < ROWS_WG * IN_F) {
            int r = tid / 3, kk = tid - r * 3;
            xs[r][kk] = x[((size_t)(b0 + r) * T_LEN + t) * IN_F + kk];
        }
        // gfx1250 prefetch of next step's x (global_prefetch_b8).
        if (tid < ROWS_WG && t + 1 < T_LEN)
            __builtin_prefetch(&x[((size_t)(b0 + tid) * T_LEN + (t + 1)) * IN_F], 0, 1);

        // ---- WMMA phase: 16x64 gate slice, K=128 in 4 chunks of 32 ----
        v8f acc[4] = {};
#pragma unroll
        for (int kc = 0; kc < 4; ++kc) {
            v16h a = *(const v16h*)&hA[(kc * 32 + lane) * 16];
#pragma unroll
            for (int n = 0; n < 4; ++n) {
                v16h b = Bf[(((n_group * 4 + n) * 4 + kc) * 32) + lane];
                acc[n] = __builtin_amdgcn_wmma_f32_16x16x32_f16(
                    false, a, false, b, (short)0, acc[n], false, false);
            }
        }

        // Store C tiles to LDS (C layout: VGPR v -> M = v + (lane/16)*8).
        {
            const int half = lane >> 4;
            const int ln   = lane & 15;
#pragma unroll
            for (int n = 0; n < 4; ++n) {
                int cidx = n_group * 64 + n * 16 + ln;
                int rb   = half * 8;
#pragma unroll
                for (int v = 0; v < 8; ++v)
                    gatesLDS[rb + v][cidx] = acc[n][v];
            }
        }
        __syncthreads();

        // ---- Elementwise LSTM update (8 rows per thread) ----
#pragma unroll
        for (int k = 0; k < 8; ++k) {
            int r = rbase + 2 * k;
            float x0 = xs[r][0], x1 = xs[r][1], x2 = xs[r][2];
            float gi = gatesLDS[r][0 * H_F + j] + x0 * wih[0][0] + x1 * wih[0][1] + x2 * wih[0][2] + bias[0];
            float gf = gatesLDS[r][1 * H_F + j] + x0 * wih[1][0] + x1 * wih[1][1] + x2 * wih[1][2] + bias[1];
            float gg = gatesLDS[r][2 * H_F + j] + x0 * wih[2][0] + x1 * wih[2][1] + x2 * wih[2][2] + bias[2];
            float go = gatesLDS[r][3 * H_F + j] + x0 * wih[3][0] + x1 * wih[3][1] + x2 * wih[3][2] + bias[3];
            float i_ = 1.f / (1.f + __expf(-gi));
            float f_ = 1.f / (1.f + __expf(-gf));
            float g_ = tanhf(gg);
            float o_ = 1.f / (1.f + __expf(-go));
            float c  = f_ * c_reg[k] + i_ * g_;
            c_reg[k] = c;
            float h  = o_ * tanhf(c);
            h_reg[k] = h;

            // Scatter f16 h into A-fragment layout (contiguous, matching Bfrag):
            // half 0 covers K 0..15, half 1 covers K 16..31.
            int kk = j & 31;
            int hh = kk >> 4, e = kk & 15;
            int lidx = hh * 16 + (r & 15);
            hA[((j >> 5) * 32 + lidx) * 16 + e] = (_Float16)h;
        }
    }

    // Final hidden state to global (f32).
#pragma unroll
    for (int k = 0; k < 8; ++k) {
        int r = rbase + 2 * k;
        hout[(size_t)(b0 + r) * H_F + j] = h_reg[k];
    }
}

// ---------------------------------------------------------------------------
// BatchNorm stats: one block per feature; fold mean/var/gamma/beta into
// scale/shift so the head kernel is a single FMA per element.
// ---------------------------------------------------------------------------
__global__ __launch_bounds__(256)
void bn_stats(const float* __restrict__ hout,
              const float* __restrict__ gamma,
              const float* __restrict__ beta,
              float* __restrict__ scale,
              float* __restrict__ shift) {
    __shared__ float s1[256], s2[256];
    int jf = blockIdx.x;
    float s = 0.f, ss = 0.f;
    for (int b = threadIdx.x; b < B_TOT; b += 256) {
        float v = hout[(size_t)b * H_F + jf];
        s += v; ss += v * v;
    }
    s1[threadIdx.x] = s; s2[threadIdx.x] = ss;
    __syncthreads();
    for (int off = 128; off > 0; off >>= 1) {
        if (threadIdx.x < off) {
            s1[threadIdx.x] += s1[threadIdx.x + off];
            s2[threadIdx.x] += s2[threadIdx.x + off];
        }
        __syncthreads();
    }
    if (threadIdx.x == 0) {
        float mean = s1[0] * (1.f / B_TOT);
        float var  = s2[0] * (1.f / B_TOT) - mean * mean;
        float inv  = rsqrtf(var + 1e-5f);
        float sc   = inv * gamma[jf];
        scale[jf]  = sc;
        shift[jf]  = beta[jf] - mean * sc;
    }
}

// ---------------------------------------------------------------------------
// Head: BN apply + LeakyReLU + fc(128->1) + sigmoid.  One wave32 per row,
// 4 features per lane, __shfl_xor reduction.
// ---------------------------------------------------------------------------
__global__ __launch_bounds__(256)
void head_out(const float* __restrict__ hout,
              const float* __restrict__ scale,
              const float* __restrict__ shift,
              const float* __restrict__ fcw,
              const float* __restrict__ fcb,
              float* __restrict__ out) {
    int lane = threadIdx.x & 31;
    int wv   = threadIdx.x >> 5;
    int b    = blockIdx.x * 8 + wv;
    float p = 0.f;
#pragma unroll
    for (int q = 0; q < 4; ++q) {
        int jf = q * 32 + lane;
        float v = hout[(size_t)b * H_F + jf] * scale[jf] + shift[jf];
        v = (v >= 0.f) ? v : 0.01f * v;   // LeakyReLU
        p += v * fcw[jf];
    }
#pragma unroll
    for (int off = 16; off > 0; off >>= 1)
        p += __shfl_xor(p, off, 32);
    if (lane == 0)
        out[b] = 1.f / (1.f + __expf(-(p + fcb[0])));
}

// ---------------------------------------------------------------------------
extern "C" void kernel_launch(void* const* d_in, const int* in_sizes, int n_in,
                              void* d_out, int out_size, void* d_ws, size_t ws_size,
                              hipStream_t stream) {
    const float* x     = (const float*)d_in[0];
    const float* Wih   = (const float*)d_in[1];
    const float* Whh   = (const float*)d_in[2];
    const float* bih   = (const float*)d_in[3];
    const float* bhh   = (const float*)d_in[4];
    const float* gamma = (const float*)d_in[5];
    const float* beta  = (const float*)d_in[6];
    const float* fcw   = (const float*)d_in[7];
    const float* fcb   = (const float*)d_in[8];
    float* out = (float*)d_out;

    char* ws = (char*)d_ws;
    _Float16* Bfrag = (_Float16*)ws;                          // 128 KB
    float* hout  = (float*)(ws + (128u << 10));               // 2 MB
    float* scale = (float*)(ws + (128u << 10) + (2u << 20));  // 512 B
    float* shift = scale + H_F;                               // 512 B

    prep_bfrag<<<256, 256, 0, stream>>>(Whh, Bfrag);
    lstm_wmma<<<B_TOT / ROWS_WG, THREADS, 0, stream>>>(x, Bfrag, Wih, bih, bhh, hout);
    bn_stats<<<H_F, 256, 0, stream>>>(hout, gamma, beta, scale, shift);
    head_out<<<B_TOT / 8, 256, 0, stream>>>(hout, scale, shift, fcw, fcb, out);
}
